// SimpleNTM_66013647339789
// MI455X (gfx1250) — matrix operand order repaired and checked by reading, not compiled, not run on hardware
//
#include <hip/hip_runtime.h>

// ---------------------------------------------------------------------------
// SimpleNTM on MI455X (gfx1250): bf16 WMMA GEMMs with fragment-order staging,
// double-buffered LDS (one barrier per K-chunk), wave32 attention kernel.
// ---------------------------------------------------------------------------

typedef __attribute__((ext_vector_type(16))) __bf16 v16bf;
typedef __attribute__((ext_vector_type(8)))  float  v8f;

#define T_DIM 256
#define B_DIM 512
#define E_DIM 256
#define H_DIM 256
#define C_DIM 256
#define N_DIM 128
#define V_DIM 256
#define O_DIM 256
#define EPSF  1e-8f

// Round-to-nearest-even f32 -> bf16 (inputs always finite here).
__device__ __forceinline__ unsigned short f32_to_bf16(float f) {
  unsigned u = __float_as_uint(f);
  return (unsigned short)((u + 0x7FFFu + ((u >> 16) & 1u)) >> 16);
}

// Pack two f32 into packed bf16x2 using one v_perm_b32 for the byte gather:
// result = { hi16(round(b)), hi16(round(a)) }.
__device__ __forceinline__ unsigned pack_bf16(float a, float b) {
  unsigned ua = __float_as_uint(a);
  unsigned ub = __float_as_uint(b);
  ua += 0x7FFFu + ((ua >> 16) & 1u);
  ub += 0x7FFFu + ((ub >> 16) & 1u);
  // v_perm_b32: sel bytes 0-3 from src1 (ua), 4-7 from src0 (ub)
  return __builtin_amdgcn_perm(ub, ua, 0x07060302u);
}

// ---------------------- weight prep: swizzle to B-fragment order -----------
// B-matrix (32x16) fragment layout used here (contiguous k order, same as A), wave32:
//   lane = (k5>>4)*16 + (n&15), j = k5&15  (k5 = k within 32-chunk)
// Storage: dst[ ((nc*KC + kc)*32 + lane)*16 + j ]  -> each lane's 16 bf16
// contiguous (32B), consecutive k-chunks contiguous (512 elems apart).

__global__ void k_swizzle_w(const float* __restrict__ src,
                            unsigned short* __restrict__ dst, int K, int N) {
  int idx = blockIdx.x * blockDim.x + threadIdx.x;
  if (idx >= K * N) return;
  int kk = idx / N, n = idx - kk * N;
  int KC = K >> 5;
  int kc = kk >> 5, k5 = kk & 31;
  int lane = ((k5 >> 4) << 4) + (n & 15);
  size_t dsti = ((((size_t)(n >> 4) * KC + kc) << 5) + lane) * 16 + (k5 & 15);
  dst[dsti] = f32_to_bf16(src[idx]);
}

// fused [Wr | Ww] (K=256, N=768) swizzled the same way
__global__ void k_swizzle_wcat(const float* __restrict__ Wr,
                               const float* __restrict__ Ww,
                               unsigned short* __restrict__ dst) {
  int idx = blockIdx.x * blockDim.x + threadIdx.x;
  if (idx >= C_DIM * 768) return;
  int kk = idx / 768, n = idx - kk * 768;
  float v = (n < V_DIM) ? Wr[kk * V_DIM + n]
                        : Ww[kk * (2 * V_DIM) + (n - V_DIM)];
  int KC = C_DIM >> 5;                      // 8
  int kc = kk >> 5, k5 = kk & 31;
  int lane = ((k5 >> 4) << 4) + (n & 15);
  size_t dsti = ((((size_t)(n >> 4) * KC + kc) << 5) + lane) * 16 + (k5 & 15);
  dst[dsti] = f32_to_bf16(v);
}

__global__ void k_build_bcat(const float* __restrict__ br,
                             const float* __restrict__ bw,
                             float* __restrict__ bcat) {
  int i = blockIdx.x * blockDim.x + threadIdx.x;
  if (i < 768) bcat[i] = (i < V_DIM) ? br[i] : bw[i - V_DIM];
}

// ------------------------------ WMMA GEMM ----------------------------------
// C[M,N] = relu?( concat(A0[M,K0], A1[M,K1]) @ W[K,N] + bias[N] )
// Macro-tile 64x64, 256 threads = 8 waves. Wave w: row block rb = w>>1
// (16 rows), col tiles tc0 = (w&1)*2 and tc0+1 -> 2 WMMAs per K-chunk.
// A staged f32->bf16 into double-buffered LDS in A-fragment order (one
// ds_store_b128 per thread per chunk; wave gather = one 32B ds read).
// B fragments loaded directly from pre-swizzled global weights (32B vector
// loads, L2-resident). One barrier per K-chunk; staging of chunk k+1
// overlaps the WMMAs of chunk k.

#define ATILE (64 * 32)

__global__ __launch_bounds__(256)
void k_gemm_bf16(const float* __restrict__ A0, int K0,
                 const float* __restrict__ A1, int K1,
                 const unsigned short* __restrict__ Wf,  // frag-order bf16
                 const float* __restrict__ bias,
                 float* __restrict__ C, int N, int relu) {
  __shared__ unsigned short ldsA[2 * ATILE];   // 8 KB, fragment order, 2 bufs
  const int K    = K0 + K1;
  const int KC   = K >> 5;
  const int tid  = threadIdx.x;
  const int lane = tid & 31;
  const int wid  = tid >> 5;
  const int mbase = blockIdx.y * 64;
  const int nbase = blockIdx.x * 64;
  const int rb  = wid >> 1;                  // wave row block (0..3)
  const int tc0 = (wid & 1) * 2;             // wave first col tile (0 / 2)
  const int hf  = lane >> 4;
  const int ml  = lane & 15;

  // A-fill assignment: thread -> (row fr, k-octet foc)
  const int fr  = tid >> 2;                  // 0..63
  const int foc = tid & 3;                   // 0..3
  const int aoff = ((((fr >> 4) << 5) + ((foc >> 1) << 4) + (fr & 15)) << 4) +
                   ((foc & 1) << 3);
  const int gcol0 = foc << 3;
  const int aread = ((rb << 5) + lane) << 4;

  const size_t nc0 = (size_t)(nbase >> 4) + tc0;
  const unsigned short* bbase0 = Wf + (((nc0 + 0) * KC) << 9) + (lane << 4);
  const unsigned short* bbase1 = Wf + (((nc0 + 1) * KC) << 9) + (lane << 4);

  // stage one 32-wide K chunk (f32 -> bf16, fragment order) into buffer `buf`
  auto stage = [&](int kb, int buf) {
    const int gc = kb + gcol0;
    const float* src = (gc < K0)
        ? A0 + (size_t)(mbase + fr) * K0 + gc
        : A1 + (size_t)(mbase + fr) * K1 + (gc - K0);
    float4 lo = *(const float4*)src;
    float4 hi = *(const float4*)(src + 4);
    uint4 pk;
    pk.x = pack_bf16(lo.x, lo.y);
    pk.y = pack_bf16(lo.z, lo.w);
    pk.z = pack_bf16(hi.x, hi.y);
    pk.w = pack_bf16(hi.z, hi.w);
    *(uint4*)&ldsA[buf * ATILE + aoff] = pk;
  };

  v8f acc0 = {}, acc1 = {};
  stage(0, 0);
  for (int kc = 0; kc < KC; ++kc) {
    __syncthreads();                         // staging of buf[kc&1] complete
    if (kc + 1 < KC) stage((kc + 1) << 5, (kc + 1) & 1);  // overlap next chunk

    v16bf fa  = *(const v16bf*)&ldsA[(kc & 1) * ATILE + aread];
    v16bf fb0 = *(const v16bf*)(bbase0 + ((size_t)kc << 9));
    v16bf fb1 = *(const v16bf*)(bbase1 + ((size_t)kc << 9));
    if (kc + 1 < KC) {                       // prefetch next B k-chunks
      __builtin_prefetch(bbase0 + (((size_t)kc + 1) << 9), 0, 3);
      __builtin_prefetch(bbase1 + (((size_t)kc + 1) << 9), 0, 3);
    }
    acc0 = __builtin_amdgcn_wmma_f32_16x16x32_bf16(
        false, fa, false, fb0, (short)0, acc0, false, false);
    acc1 = __builtin_amdgcn_wmma_f32_16x16x32_bf16(
        false, fa, false, fb1, (short)0, acc1, false, false);
  }

  // ---- C/D layout: VGPR rr -> M = hf*8 + rr, N = lane&15
  const int col0 = nbase + (tc0 << 4) + ml;
  const float b0 = bias ? bias[col0] : 0.f;
  const float b1 = bias ? bias[col0 + 16] : 0.f;
  #pragma unroll
  for (int rr = 0; rr < 8; ++rr) {
    const int row = mbase + (rb << 4) + (hf << 3) + rr;
    float* crow = C + (size_t)row * N;
    float v0 = acc0[rr] + b0;
    float v1 = acc1[rr] + b1;
    if (relu) { v0 = fmaxf(v0, 0.f); v1 = fmaxf(v1, 0.f); }
    crow[col0]      = v0;
    crow[col0 + 16] = v1;
  }
}

// --------------------- attention / memory step kernel ----------------------
// One workgroup per batch b (256 threads = 8 waves): cosine scores for read
// key k and write key kw, dual softmax over N=128 slots, read r = w @ mem
// (pre-update), then mem += ww (outer) v. mem is L2-resident (64MB < 192MB).

__global__ __launch_bounds__(256)
void k_mem_step(const float* __restrict__ kkwv,   // [B, 768] = k | kw | v
                float* __restrict__ mem,          // [B, N, V]
                float* __restrict__ readout) {    // [B, V]
  __shared__ float sk[V_DIM], skw[V_DIM], sv[V_DIM];
  __shared__ float swr[N_DIM], sww[N_DIM];
  __shared__ float scal[2];
  const int b    = blockIdx.x;
  const int tid  = threadIdx.x;
  const int lane = tid & 31;
  const int wid  = tid >> 5;

  const float* kb = kkwv + (size_t)b * 768;
  sk[tid]  = kb[tid];
  skw[tid] = kb[256 + tid];
  sv[tid]  = kb[512 + tid];
  __syncthreads();

  if (tid == 0) {
    float a = 0.f, c = 0.f;
    for (int j = 0; j < V_DIM; ++j) { a += sk[j] * sk[j]; c += skw[j] * skw[j]; }
    scal[0] = fmaxf(sqrtf(a), EPSF);
    scal[1] = fmaxf(sqrtf(c), EPSF);
  }
  __syncthreads();
  const float kn = scal[0], kwn = scal[1];

  float* memb = mem + (size_t)b * N_DIM * V_DIM;

  // cosine scores: wave wid owns slots n = wid*16 .. wid*16+15
  for (int i = 0; i < 16; ++i) {
    int n = wid * 16 + i;
    const float* mr = memb + (size_t)n * V_DIM;
    float dr = 0.f, dw = 0.f, nn = 0.f;
    for (int j = lane; j < V_DIM; j += 32) {
      float m = mr[j];
      dr += sk[j] * m;
      dw += skw[j] * m;
      nn += m * m;
    }
    #pragma unroll
    for (int off = 16; off > 0; off >>= 1) {
      dr += __shfl_xor(dr, off, 32);
      dw += __shfl_xor(dw, off, 32);
      nn += __shfl_xor(nn, off, 32);
    }
    if (lane == 0) {
      float mn = fmaxf(sqrtf(nn), EPSF);
      swr[n] = dr / (kn * mn);
      sww[n] = dw / (kwn * mn);
    }
  }
  __syncthreads();

  // dual softmax over N=128 (tiny; serial on one lane)
  if (tid == 0) {
    float m1 = -1e30f, m2 = -1e30f;
    for (int n = 0; n < N_DIM; ++n) { m1 = fmaxf(m1, swr[n]); m2 = fmaxf(m2, sww[n]); }
    float s1 = 0.f, s2 = 0.f;
    for (int n = 0; n < N_DIM; ++n) {
      float e1 = __expf(swr[n] - m1); swr[n] = e1; s1 += e1;
      float e2 = __expf(sww[n] - m2); sww[n] = e2; s2 += e2;
    }
    scal[0] = 1.f / s1;
    scal[1] = 1.f / s2;
  }
  __syncthreads();
  const float i1 = scal[0], i2 = scal[1];

  // read (pre-update mem) + in-place update; thread = column, coalesced rows
  float racc = 0.f;
  const float vj = sv[tid];
  for (int n = 0; n < N_DIM; ++n) {
    float* p = memb + (size_t)n * V_DIM + tid;
    float m = *p;
    racc += (swr[n] * i1) * m;
    *p = m + (sww[n] * i2) * vj;
  }
  readout[(size_t)b * V_DIM + tid] = racc;
}

// ------------------------------- launcher ----------------------------------

extern "C" void kernel_launch(void* const* d_in, const int* in_sizes, int n_in,
                              void* d_out, int out_size, void* d_ws, size_t ws_size,
                              hipStream_t stream) {
  const float* x         = (const float*)d_in[0];
  const float* W1        = (const float*)d_in[1];
  const float* b1        = (const float*)d_in[2];
  const float* W2        = (const float*)d_in[3];
  const float* b2        = (const float*)d_in[4];
  const float* Wr        = (const float*)d_in[5];
  const float* br        = (const float*)d_in[6];
  const float* Ww        = (const float*)d_in[7];
  const float* bw        = (const float*)d_in[8];
  const float* Wo        = (const float*)d_in[9];
  const float* bo        = (const float*)d_in[10];
  const float* init_read = (const float*)d_in[11];
  const float* mem0      = (const float*)d_in[12];
  float* out = (float*)d_out;

  char* ws = (char*)d_ws;
  size_t off = 0;
  auto alloc = [&](size_t bytes) -> char* {
    char* p = ws + off;
    off += (bytes + 255) & ~(size_t)255;
    return p;
  };
  unsigned short* w1f  = (unsigned short*)alloc((size_t)512 * 256 * 2);
  unsigned short* w2f  = (unsigned short*)alloc((size_t)256 * 256 * 2);
  unsigned short* wcat = (unsigned short*)alloc((size_t)256 * 768 * 2);
  unsigned short* wof  = (unsigned short*)alloc((size_t)512 * 256 * 2);
  float* bcat = (float*)alloc(768 * 4);
  float* h    = (float*)alloc((size_t)B_DIM * H_DIM * 4);
  float* co   = (float*)alloc((size_t)B_DIM * C_DIM * 4);
  float* kkwv = (float*)alloc((size_t)B_DIM * 768 * 4);
  float* rdo  = (float*)alloc((size_t)B_DIM * V_DIM * 4);
  float* memb = (float*)alloc((size_t)B_DIM * N_DIM * V_DIM * 4);

  // per-call prep (deterministic): fragment-swizzled bf16 weights + state reset
  k_swizzle_w<<<(512 * 256 + 255) / 256, 256, 0, stream>>>(W1, w1f, 512, 256);
  k_swizzle_w<<<(256 * 256 + 255) / 256, 256, 0, stream>>>(W2, w2f, 256, 256);
  k_swizzle_w<<<(512 * 256 + 255) / 256, 256, 0, stream>>>(Wo, wof, 512, 256);
  k_swizzle_wcat<<<(256 * 768 + 255) / 256, 256, 0, stream>>>(Wr, Ww, wcat);
  k_build_bcat<<<3, 256, 0, stream>>>(br, bw, bcat);
  hipMemcpyAsync(memb, mem0, (size_t)B_DIM * N_DIM * V_DIM * 4,
                 hipMemcpyDeviceToDevice, stream);
  hipMemcpyAsync(rdo, init_read, (size_t)B_DIM * V_DIM * 4,
                 hipMemcpyDeviceToDevice, stream);

  for (int t = 0; t < T_DIM; ++t) {
    const float* xt = x + (size_t)t * B_DIM * E_DIM;
    // h = relu([xt | read] @ W1 + b1)
    k_gemm_bf16<<<dim3(H_DIM / 64, B_DIM / 64), 256, 0, stream>>>(
        xt, E_DIM, rdo, V_DIM, w1f, b1, h, H_DIM, 1);
    // co = h @ W2 + b2
    k_gemm_bf16<<<dim3(C_DIM / 64, B_DIM / 64), 256, 0, stream>>>(
        h, H_DIM, nullptr, 0, w2f, b2, co, C_DIM, 0);
    // [k | kw | v] = co @ [Wr | Ww] + [br | bw]
    k_gemm_bf16<<<dim3(768 / 64, B_DIM / 64), 256, 0, stream>>>(
        co, C_DIM, nullptr, 0, wcat, bcat, kkwv, 768, 0);
    // attention read + memory write
    k_mem_step<<<B_DIM, 256, 0, stream>>>(kkwv, memb, rdo);
    // out[t] = [co | r] @ Wo + bo
    k_gemm_bf16<<<dim3(O_DIM / 64, B_DIM / 64), 256, 0, stream>>>(
        co, C_DIM, rdo, V_DIM, wof, bo, out + (size_t)t * B_DIM * O_DIM, O_DIM, 0);
  }
}
